// KSSMBlock_28707561406893
// MI455X (gfx1250) — compile-verified, matrix operand order checked by reading, not run on hardware
//
#include <hip/hip_runtime.h>
#include <math.h>

// ---- problem constants (from reference) ----
#define BB 2
#define LL 2048
#define DM_ 1024
#define DIN 2048
#define HH 16
#define HD 128
#define KCONV 4
#define NGROUPS 16
#define MM (BB*LL)             // 4096 tokens
#define NPROJ (3*DIN + 2*HH)   // 6176
#define GATING_C 7.6246189861593985f   // log(2048)

typedef __bf16 bf16;
typedef __attribute__((ext_vector_type(8)))  __bf16 v8bf;
typedef __attribute__((ext_vector_type(16))) __bf16 v16bf;
typedef __attribute__((ext_vector_type(8)))  float  v8f;
typedef __attribute__((ext_vector_type(4)))  int    v4i;

// gfx1250 async global->LDS path (ASYNCcnt-tracked), guarded so the file
// always compiles even if this toolchain doesn't expose the builtins.
#if defined(__AMDGCN__) && __has_builtin(__builtin_amdgcn_global_load_async_to_lds_b128) && __has_builtin(__builtin_amdgcn_s_wait_asynccnt)
#define ASYNC_LDS 1
#else
#define ASYNC_LDS 0
#endif

// builtin expects int4-vector pointers (per clang diagnostic), global/LDS AS
#define GAS1(p) ((__attribute__((address_space(1))) v4i*)(p))
#define LAS3(p) ((__attribute__((address_space(3))) v4i*)(p))

__device__ __forceinline__ float sigm_f(float x)  { return 1.0f / (1.0f + __expf(-x)); }
__device__ __forceinline__ float silu_f(float x)  { return x / (1.0f + __expf(-x)); }
__device__ __forceinline__ float softplus_f(float x){ return x > 20.0f ? x : log1pf(__expf(x)); }

// ---------------------------------------------------------------------------
// Generic cast fp32 -> bf16
// ---------------------------------------------------------------------------
__global__ void cast_f32_bf16(const float* __restrict__ in, bf16* __restrict__ out, size_t n) {
  size_t i = (size_t)blockIdx.x * blockDim.x + threadIdx.x;
  if (i < n) out[i] = (bf16)in[i];
}

// ---------------------------------------------------------------------------
// RMSNorm: xn = x * rsqrt(mean(x^2)+1e-6) * w   (one block per row of DM_)
// ---------------------------------------------------------------------------
__global__ __launch_bounds__(256) void rmsnorm_kernel(const float* __restrict__ x,
                                                      const float* __restrict__ w,
                                                      bf16* __restrict__ out) {
  int m = blockIdx.x, tid = threadIdx.x;
  __shared__ float sh[256];
  const float* xr = x + (size_t)m * DM_;
  float ss = 0.f;
  for (int c = tid; c < DM_; c += 256) { float v = xr[c]; ss += v * v; }
  sh[tid] = ss; __syncthreads();
  for (int o = 128; o > 0; o >>= 1) { if (tid < o) sh[tid] += sh[tid + o]; __syncthreads(); }
  float scale = rsqrtf(sh[0] / (float)DM_ + 1e-6f);
  for (int c = tid; c < DM_; c += 256)
    out[(size_t)m * DM_ + c] = (bf16)(xr[c] * scale * w[c]);
}

// ---------------------------------------------------------------------------
// bf16 WMMA GEMM:  C[M,N] = A[M,K] @ B[N,K]^T (+ bias[N]) (+ resid[M,N])
// WG tile 128x128, 8 waves (wave32), wave tile 32x64, BK=32, double-buffered
// LDS. Tile fills use gfx1250 global_load_async_to_lds_b128 when available.
// ---------------------------------------------------------------------------
__global__ __launch_bounds__(256) void gemm_bf16_wmma(const bf16* __restrict__ A,
                                                      const bf16* __restrict__ Bm,
                                                      float* __restrict__ C,
                                                      const float* __restrict__ bias,
                                                      const float* __restrict__ resid,
                                                      int M, int N, int K) {
  __shared__ __align__(16) bf16 As[2][128][40];   // +8 bf16 pad -> 80B row stride
  __shared__ __align__(16) bf16 Bs[2][128][40];

  const int mBase = blockIdx.y * 128;
  const int nBase = blockIdx.x * 128;
  const int tid  = threadIdx.x;
  const int r    = tid >> 1;          // 0..127 tile row to fill
  const int kc   = (tid & 1) << 4;    // 0 or 16 (bf16 elements)
  const int wid  = tid >> 5, lane = tid & 31;
  const int wm   = wid & 3;           // 4 waves along M (32 rows each)
  const int wn   = wid >> 2;          // 2 waves along N (64 cols each)
  const int hi   = lane >> 4, ln = lane & 15;

  v8f acc[2][4];
  for (int i = 0; i < 2; ++i)
    for (int j = 0; j < 4; ++j)
      acc[i][j] = (v8f){0.f,0.f,0.f,0.f,0.f,0.f,0.f,0.f};

  auto fill = [&](int buf, int k0) {
#if ASYNC_LDS
    int gr = mBase + r;
    if (gr < M) {
      const bf16* g = A + (size_t)gr * K + k0 + kc;
      __builtin_amdgcn_global_load_async_to_lds_b128(GAS1(g),     LAS3(&As[buf][r][kc]),     0, 0);
      __builtin_amdgcn_global_load_async_to_lds_b128(GAS1(g + 8), LAS3(&As[buf][r][kc + 8]), 0, 0);
    } else {
      uint4 z = {0,0,0,0};
      uint4* sa = reinterpret_cast<uint4*>(&As[buf][r][kc]); sa[0] = z; sa[1] = z;
    }
    int gn = nBase + r;
    if (gn < N) {
      const bf16* g = Bm + (size_t)gn * K + k0 + kc;
      __builtin_amdgcn_global_load_async_to_lds_b128(GAS1(g),     LAS3(&Bs[buf][r][kc]),     0, 0);
      __builtin_amdgcn_global_load_async_to_lds_b128(GAS1(g + 8), LAS3(&Bs[buf][r][kc + 8]), 0, 0);
    } else {
      uint4 z = {0,0,0,0};
      uint4* sb = reinterpret_cast<uint4*>(&Bs[buf][r][kc]); sb[0] = z; sb[1] = z;
    }
#else
    uint4 a0 = {0,0,0,0}, a1 = a0, b0 = a0, b1 = a0;
    int gr = mBase + r;
    if (gr < M) {
      const uint4* g = reinterpret_cast<const uint4*>(A + (size_t)gr * K + k0 + kc);
      a0 = g[0]; a1 = g[1];
    }
    int gn = nBase + r;
    if (gn < N) {
      const uint4* g = reinterpret_cast<const uint4*>(Bm + (size_t)gn * K + k0 + kc);
      b0 = g[0]; b1 = g[1];
    }
    uint4* sa = reinterpret_cast<uint4*>(&As[buf][r][kc]); sa[0] = a0; sa[1] = a1;
    uint4* sb = reinterpret_cast<uint4*>(&Bs[buf][r][kc]); sb[0] = b0; sb[1] = b1;
#endif
  };

  auto ldfrag = [&](const bf16 (*T)[40], int row) -> v16bf {
    v8bf lo = *reinterpret_cast<const v8bf*>(&T[row][8 * hi]);
    v8bf hp = *reinterpret_cast<const v8bf*>(&T[row][16 + 8 * hi]);
    return __builtin_shufflevector(lo, hp, 0,1,2,3,4,5,6,7,8,9,10,11,12,13,14,15);
  };

  const int nStages = K >> 5;
  fill(0, 0);
#if ASYNC_LDS
  __builtin_amdgcn_s_wait_asynccnt(0);
#endif
  __syncthreads();

  for (int s = 0; s < nStages; ++s) {
    const int buf = s & 1;
    if (s + 1 < nStages) fill(buf ^ 1, (s + 1) << 5);
    if (s + 2 < nStages) {  // gfx1250 global_prefetch_b8 for stage-after-next
      int k2 = (s + 2) << 5;
      int gr = mBase + r, gn = nBase + r;
      if (gr < M) __builtin_prefetch(A  + (size_t)gr * K + k2 + kc, 0, 1);
      if (gn < N) __builtin_prefetch(Bm + (size_t)gn * K + k2 + kc, 0, 1);
    }
    v16bf afr[2], bfr[4];
    for (int mt = 0; mt < 2; ++mt) afr[mt] = ldfrag(As[buf], wm * 32 + mt * 16 + ln);
    for (int nt = 0; nt < 4; ++nt) bfr[nt] = ldfrag(Bs[buf], wn * 64 + nt * 16 + ln);
    for (int mt = 0; mt < 2; ++mt)
      for (int nt = 0; nt < 4; ++nt)
        acc[mt][nt] = __builtin_amdgcn_wmma_f32_16x16x32_bf16(
            false, afr[mt], false, bfr[nt], (short)0, acc[mt][nt], false, false);
#if ASYNC_LDS
    __builtin_amdgcn_s_wait_asynccnt(0);
#endif
    __syncthreads();
  }

  // store: VGPR j holds M = j + 8*hi within the 16x16 tile; N = ln
  for (int mt = 0; mt < 2; ++mt)
    for (int nt = 0; nt < 4; ++nt)
      for (int j = 0; j < 8; ++j) {
        int gm = mBase + wm * 32 + mt * 16 + 8 * hi + j;
        int gn = nBase + wn * 64 + nt * 16 + ln;
        if (gm < M && gn < N) {
          float v = acc[mt][nt][j];
          if (bias)  v += bias[gn];
          if (resid) v += resid[(size_t)gm * N + gn];
          C[(size_t)gm * N + gn] = v;
        }
      }
}

// ---------------------------------------------------------------------------
// Causal depthwise conv (K=4) over x_gate slice of proj, + SiLU.
// ---------------------------------------------------------------------------
__global__ void conv_silu_kernel(const float* __restrict__ proj,
                                 const float* __restrict__ cw,
                                 const float* __restrict__ cb,
                                 float* __restrict__ xconv,
                                 bf16* __restrict__ xconvbf) {
  size_t idx = (size_t)blockIdx.x * blockDim.x + threadIdx.x;
  if (idx >= (size_t)MM * DIN) return;
  int c = (int)(idx & (DIN - 1));
  int m = (int)(idx >> 11);
  int b = m >> 11, l = m & (LL - 1);   // L == 2048
  float acc = cb[c];
  #pragma unroll
  for (int k = 0; k < KCONV; ++k) {
    int lp = l - (KCONV - 1) + k;
    if (lp >= 0)
      acc += proj[((size_t)(b * LL + lp)) * NPROJ + DIN + c] * cw[c * KCONV + k];
  }
  float s = silu_f(acc);
  xconv[idx] = s;
  xconvbf[idx] = (bf16)s;
}

// ---------------------------------------------------------------------------
// Pack the 7 small projection weights into one [128 x DIN] bf16 matrix.
// rows: 0-31 dyn, 32-47 selB, 48-63 selC, 64-79 seldt, 80-95 beta, 96-111 rg, 112-127 ug
// ---------------------------------------------------------------------------
__global__ void pack_small_kernel(const float* dynw, const float* sB, const float* sC,
                                  const float* sdt, const float* bw, const float* rw,
                                  const float* uw, bf16* out) {
  size_t idx = (size_t)blockIdx.x * blockDim.x + threadIdx.x;
  if (idx >= (size_t)128 * DIN) return;
  int row = (int)(idx / DIN), c = (int)(idx % DIN);
  float v;
  if      (row <  32) v = dynw[(size_t)row * DIN + c];
  else if (row <  48) v = sB [(size_t)(row - 32) * DIN + c];
  else if (row <  64) v = sC [(size_t)(row - 48) * DIN + c];
  else if (row <  80) v = sdt[(size_t)(row - 64) * DIN + c];
  else if (row <  96) v = bw [(size_t)(row - 80) * DIN + c];
  else if (row < 112) v = rw [(size_t)(row - 96) * DIN + c];
  else                v = uw [(size_t)(row - 112) * DIN + c];
  out[idx] = (bf16)v;
}

// ---------------------------------------------------------------------------
// Pointwise gating + K/Q l2-normalization. One block per (m,h), 128 lanes = d.
// ---------------------------------------------------------------------------
__global__ __launch_bounds__(128) void pointwise_kernel(
    const float* __restrict__ proj, float* __restrict__ q /*in-out*/,
    const float* __restrict__ sm_all, float* __restrict__ kn,
    const float* dyn_b, const float* dt_log, const float* beta_b,
    const float* rg_b, const float* ug_b, const float* sg_w, const float* ema,
    float* abuf, float* bbuf, float* betabuf, float* vgbuf) {
  int m = blockIdx.x >> 4, h = blockIdx.x & 15, d = threadIdx.x;
  int l = m & (LL - 1);
  const float* sm = sm_all + (size_t)m * 128;
  float selB = sm[32 + h], selC = sm[48 + h];
  float kv = proj[(size_t)m * NPROJ + 2 * DIN + h * HD + d] * selB;
  float qv = q[(size_t)m * DIN + h * HD + d] * selC;

  __shared__ float red[4][2];
  float s0 = kv * kv, s1 = qv * qv;
  for (int o = 16; o > 0; o >>= 1) {
    s0 += __shfl_xor(s0, o, 32);
    s1 += __shfl_xor(s1, o, 32);
  }
  int lane = d & 31, w = d >> 5;
  if (lane == 0) { red[w][0] = s0; red[w][1] = s1; }
  __syncthreads();
  float kn2 = red[0][0] + red[1][0] + red[2][0] + red[3][0];
  float qn2 = red[0][1] + red[1][1] + red[2][1] + red[3][1];
  float kinv = 1.0f / fmaxf(sqrtf(kn2), 1e-12f);
  float qinv = 1.0f / fmaxf(sqrtf(qn2), 1e-12f);
  kn[(size_t)m * DIN + h * HD + d] = kv * kinv;
  q [(size_t)m * DIN + h * HD + d] = qv * qinv;

  if (d == 0) {
    float alpha = softplus_f(sm[h] + dyn_b[h]);
    float freq  = __expf(-((float)h / (float)HH) * logf(10000.0f));
    float omega = sm[16 + h] + dyn_b[16 + h] + (float)l * freq;
    float dt0   = softplus_f(dt_log[h]);
    float magc  = sqrtf(alpha * alpha + omega * omega);
    float dt    = dt0 / (1.0f + dt0 * magc) + softplus_f(sm[64 + h]);
    float beta  = sigm_f(sm[80 + h] + beta_b[h]);
    float rg    = sigm_f(sm[96 + h] + rg_b[h]);
    float ss = 0.f;
    for (int j = 0; j < HH; ++j) ss += ema[j] * sg_w[h * HH + j];
    float ug  = sigm_f(sm[112 + h] + ug_b[h] + ss);
    float hdt = 0.5f * dt;
    float den_re = 1.f + hdt * alpha, den_im = -hdt * omega;
    float num_re = 1.f - hdt * alpha, num_im =  hdt * omega;
    float d2 = den_re * den_re + den_im * den_im;
    float a_ = (num_re * den_re + num_im * den_im) / d2;
    float b_ = (num_im * den_re - num_re * den_im) / d2;
    float mag = fminf(fmaxf(sqrtf(a_ * a_ + b_ * b_), 1e-8f), 1.f - 1e-6f);
    float nm  = __powf(mag, GATING_C * rg);
    float sc  = nm / mag; a_ *= sc; b_ *= sc;
    float vp  = sqrtf(fminf(fmaxf(1.f - nm * nm, 0.f), 1.f));
    float g   = vp * ug;
    size_t o16 = (size_t)m * HH + h;
    abuf[o16] = a_; bbuf[o16] = b_; betabuf[o16] = beta;
    vgbuf[o16 * 2 + 0] = proj[(size_t)m * NPROJ + 3 * DIN + h * 2 + 0] * g;
    vgbuf[o16 * 2 + 1] = proj[(size_t)m * NPROJ + 3 * DIN + h * 2 + 1] * g;
  }
}

// ---------------------------------------------------------------------------
// Sequential delta-rule scan with fused Q retrieval. One block per (b,h),
// 128 lanes each own one state dim d. Ys never materialized.
// ---------------------------------------------------------------------------
__global__ __launch_bounds__(128) void scan_kernel(
    const float* __restrict__ abuf, const float* __restrict__ bbuf,
    const float* __restrict__ betabuf, const float* __restrict__ vg,
    const float* __restrict__ kn, const float* __restrict__ qn,
    float* __restrict__ ret) {
  int bh = blockIdx.x, b = bh >> 4, h = bh & 15;
  int d = threadIdx.x, lane = d & 31, w = d >> 5;
  __shared__ float red[4][4];
  float S0 = 0.f, S1 = 0.f;
  for (int t = 0; t < LL; ++t) {
    size_t m = (size_t)b * LL + t;
    float a   = abuf[m * HH + h];
    float bb  = bbuf[m * HH + h];
    float bet = betabuf[m * HH + h];
    float kd  = kn[m * DIN + h * HD + d];
    float qd  = qn[m * DIN + h * HD + d];
    float r0 =  a * S0 + bb * S1;
    float r1 = -bb * S0 + a * S1;
    float p0 = r0 * kd, p1 = r1 * kd;
    for (int o = 16; o > 0; o >>= 1) { p0 += __shfl_xor(p0, o, 32); p1 += __shfl_xor(p1, o, 32); }
    if (lane == 0) { red[w][0] = p0; red[w][1] = p1; }
    __syncthreads();
    p0 = red[0][0] + red[1][0] + red[2][0] + red[3][0];
    p1 = red[0][1] + red[1][1] + red[2][1] + red[3][1];
    __syncthreads();
    float v0 = vg[(m * HH + h) * 2 + 0], v1 = vg[(m * HH + h) * 2 + 1];
    S0 = r0 + bet * (v0 - p0) * kd;
    S1 = r1 + bet * (v1 - p1) * kd;
    float y0 = S0 * qd, y1 = S1 * qd;
    for (int o = 16; o > 0; o >>= 1) { y0 += __shfl_xor(y0, o, 32); y1 += __shfl_xor(y1, o, 32); }
    if (lane == 0) { red[w][2] = y0; red[w][3] = y1; }
    __syncthreads();
    if (d == 0) {
      ret[m * 32 + h * 2 + 0] = red[0][2] + red[1][2] + red[2][2] + red[3][2];
      ret[m * 32 + h * 2 + 1] = red[0][3] + red[1][3] + red[2][3] + red[3][3];
    }
    __syncthreads();
  }
}

// ---------------------------------------------------------------------------
// GroupNorm stats over (channels-in-group, L) per (b,g).
// ---------------------------------------------------------------------------
__global__ __launch_bounds__(256) void gn_stats_kernel(const float* __restrict__ y,
                                                       float* __restrict__ stats) {
  int bg = blockIdx.x, b = bg >> 4, g = bg & 15;
  __shared__ float sh[256], sh2[256];
  float s = 0.f, s2 = 0.f;
  const int CG = DIN / NGROUPS;   // 128
  for (int i = threadIdx.x; i < CG * LL; i += 256) {
    int c = i >> 11, l = i & (LL - 1);
    float v = y[((size_t)b * LL + l) * DIN + g * CG + c];
    s += v; s2 += v * v;
  }
  sh[threadIdx.x] = s; sh2[threadIdx.x] = s2; __syncthreads();
  for (int o = 128; o > 0; o >>= 1) {
    if (threadIdx.x < o) { sh[threadIdx.x] += sh[threadIdx.x + o]; sh2[threadIdx.x] += sh2[threadIdx.x + o]; }
    __syncthreads();
  }
  if (threadIdx.x == 0) {
    float n = (float)(CG * LL);
    float mu = sh[0] / n;
    float var = sh2[0] / n - mu * mu;
    stats[bg * 2 + 0] = mu;
    stats[bg * 2 + 1] = rsqrtf(var + 1e-5f);
  }
}

// ---------------------------------------------------------------------------
// Apply GN affine, gate by silu(z), add D*x_conv -> bf16 for final GEMM.
// ---------------------------------------------------------------------------
__global__ void gn_apply_kernel(const float* __restrict__ y, const float* __restrict__ stats,
                                const float* __restrict__ gw, const float* __restrict__ gb,
                                const float* __restrict__ proj, const float* __restrict__ xconv,
                                const float* __restrict__ Dp, bf16* __restrict__ yf) {
  size_t idx = (size_t)blockIdx.x * blockDim.x + threadIdx.x;
  if (idx >= (size_t)MM * DIN) return;
  int c = (int)(idx & (DIN - 1));
  size_t m = idx >> 11;
  int b = (int)(m >> 11);
  int g = c >> 7;
  float mu = stats[(b * 16 + g) * 2], rs = stats[(b * 16 + g) * 2 + 1];
  float v = (y[idx] - mu) * rs * gw[c] + gb[c];
  float z = proj[m * NPROJ + c];
  float out = v * silu_f(z) + Dp[c] * xconv[idx];
  yf[idx] = (bf16)out;
}

// ---------------------------------------------------------------------------
extern "C" void kernel_launch(void* const* d_in, const int* in_sizes, int n_in,
                              void* d_out, int out_size, void* d_ws, size_t ws_size,
                              hipStream_t stream) {
  (void)in_sizes; (void)n_in; (void)out_size; (void)ws_size;
  const float* x         = (const float*)d_in[0];
  const float* norm_w    = (const float*)d_in[1];
  const float* in_proj_w = (const float*)d_in[2];
  const float* in_proj_b = (const float*)d_in[3];
  const float* conv_w    = (const float*)d_in[4];
  const float* conv_b    = (const float*)d_in[5];
  const float* dyn_w     = (const float*)d_in[6];
  const float* dyn_b     = (const float*)d_in[7];
  const float* dt_log    = (const float*)d_in[8];
  const float* selB_w    = (const float*)d_in[9];
  const float* selC_w    = (const float*)d_in[10];
  const float* seldt_w   = (const float*)d_in[11];
  const float* beta_w    = (const float*)d_in[12];
  const float* beta_b    = (const float*)d_in[13];
  const float* rg_w      = (const float*)d_in[14];
  const float* rg_b      = (const float*)d_in[15];
  const float* ug_w      = (const float*)d_in[16];
  const float* ug_b      = (const float*)d_in[17];
  const float* sg_w      = (const float*)d_in[18];
  const float* ema       = (const float*)d_in[19];
  const float* Q_w       = (const float*)d_in[20];
  const float* readout_w = (const float*)d_in[21];
  const float* out_w     = (const float*)d_in[22];
  const float* gn_w      = (const float*)d_in[23];
  const float* gn_b      = (const float*)d_in[24];
  const float* D_param   = (const float*)d_in[25];
  float* out = (float*)d_out;

  // bump allocator over d_ws
  char* base = (char*)d_ws;
  size_t off = 0;
  auto alloc = [&](size_t bytes) -> char* {
    off = (off + 255) & ~(size_t)255;
    char* p = base + off;
    off += bytes;
    return p;
  };
  bf16*  xnbf    = (bf16*) alloc((size_t)MM * DM_ * 2);
  bf16*  wbf     = (bf16*) alloc((size_t)NPROJ * DM_ * 2);   // reused for all weight casts
  bf16*  wsmall  = (bf16*) alloc((size_t)128 * DIN * 2);
  float* proj    = (float*)alloc((size_t)MM * NPROJ * 4);
  float* xconv   = (float*)alloc((size_t)MM * DIN * 4);
  bf16*  xconvbf = (bf16*) alloc((size_t)MM * DIN * 2);
  float* smallb  = (float*)alloc((size_t)MM * 128 * 4);
  float* qbuf    = (float*)alloc((size_t)MM * DIN * 4);
  float* knbuf   = (float*)alloc((size_t)MM * DIN * 4);
  float* abuf    = (float*)alloc((size_t)MM * HH * 4);
  float* bbuf    = (float*)alloc((size_t)MM * HH * 4);
  float* betabuf = (float*)alloc((size_t)MM * HH * 4);
  float* vgbuf   = (float*)alloc((size_t)MM * HH * 2 * 4);
  float* retbuf  = (float*)alloc((size_t)MM * 32 * 4);
  bf16*  retbf   = (bf16*) alloc((size_t)MM * 32 * 2);
  float* ybuf    = (float*)alloc((size_t)MM * DIN * 4);
  bf16*  yfbf    = (bf16*) alloc((size_t)MM * DIN * 2);
  float* stats   = (float*)alloc((size_t)BB * NGROUPS * 2 * 4);

  auto castN = [&](const float* src, bf16* dst, size_t n) {
    cast_f32_bf16<<<dim3((unsigned)((n + 255) / 256)), 256, 0, stream>>>(src, dst, n);
  };
  auto gemm = [&](const bf16* A, const bf16* Bw, float* C, const float* bias,
                  const float* resid, int M, int N, int K) {
    dim3 grid((N + 127) / 128, (M + 127) / 128);
    gemm_bf16_wmma<<<grid, 256, 0, stream>>>(A, Bw, C, bias, resid, M, N, K);
  };

  // 1) RMSNorm -> bf16 activations
  rmsnorm_kernel<<<MM, 256, 0, stream>>>(x, norm_w, xnbf);

  // 2) in_proj GEMM: proj[M,6176] = xn @ in_proj_w^T + b
  castN(in_proj_w, wbf, (size_t)NPROJ * DM_);
  gemm(xnbf, wbf, proj, in_proj_b, nullptr, MM, NPROJ, DM_);

  // 3) causal depthwise conv + SiLU
  conv_silu_kernel<<<(MM * DIN) / 256, 256, 0, stream>>>(proj, conv_w, conv_b, xconv, xconvbf);

  // 4) packed small projections: [M,128] = x_conv @ Wsmall^T
  pack_small_kernel<<<(128 * DIN) / 256, 256, 0, stream>>>(
      dyn_w, selB_w, selC_w, seldt_w, beta_w, rg_w, ug_w, wsmall);
  gemm(xconvbf, wsmall, smallb, nullptr, nullptr, MM, 128, DIN);

  // 5) Q GEMM: qraw[M,2048] = x_conv @ Q_w^T
  castN(Q_w, wbf, (size_t)DIN * DIN);
  gemm(xconvbf, wbf, qbuf, nullptr, nullptr, MM, DIN, DIN);

  // 6) gating scalars + K/Q l2-norm
  pointwise_kernel<<<MM * HH, 128, 0, stream>>>(
      proj, qbuf, smallb, knbuf, dyn_b, dt_log, beta_b, rg_b, ug_b, sg_w, ema,
      abuf, bbuf, betabuf, vgbuf);

  // 7) sequential scan (fused retrieval)
  scan_kernel<<<BB * HH, 128, 0, stream>>>(abuf, bbuf, betabuf, vgbuf, knbuf, qbuf, retbuf);

  // 8) readout GEMM: y[M,2048] = retrieved @ readout_w^T  (K=32)
  castN(retbuf, retbf, (size_t)MM * 32);
  castN(readout_w, wbf, (size_t)DIN * 32);
  gemm(retbf, wbf, ybuf, nullptr, nullptr, MM, DIN, 32);

  // 9) GroupNorm over (C/G, L) + silu(z) gate + D*x_conv
  gn_stats_kernel<<<BB * NGROUPS, 256, 0, stream>>>(ybuf, stats);
  gn_apply_kernel<<<(MM * DIN) / 256, 256, 0, stream>>>(
      ybuf, stats, gn_w, gn_b, proj, xconv, D_param, yfbf);

  // 10) out GEMM with residual: out[M,1024] = x + yfinal @ out_w^T
  castN(out_w, wbf, (size_t)DM_ * DIN);
  gemm(yfbf, wbf, out, nullptr, x, MM, DM_, DIN);
}
